// FixedVectorizedSplatFlowLayer_67688684585067
// MI455X (gfx1250) — matrix operand order repaired and checked by reading, not compiled, not run on hardware
//
#include <hip/hip_runtime.h>
#include <hip/hip_bf16.h>

typedef __bf16 v16bf __attribute__((ext_vector_type(16)));
typedef float  v8f   __attribute__((ext_vector_type(8)));
typedef int    v4i   __attribute__((ext_vector_type(4)));

#define LDK 40   // 32 + 8 pad; 40*2B = 80B row stride (multiple of 16B -> b128 ok)

#if __has_builtin(__builtin_amdgcn_global_load_async_to_lds_b128) && \
    __has_builtin(__builtin_amdgcn_s_wait_asynccnt)
#define ASYNC_LDS 1
#else
#define ASYNC_LDS 0
#endif

#define AS1 __attribute__((address_space(1)))
#define AS3 __attribute__((address_space(3)))

// async (or sync-fallback) copy of 8 bf16 (16B) global -> LDS
__device__ __forceinline__ void g2l_b128(const __bf16* g, __bf16* l) {
#if ASYNC_LDS
    __builtin_amdgcn_global_load_async_to_lds_b128(
        (AS1 v4i*)(unsigned long long)g, (AS3 v4i*)(void*)l, 0, 0);
#else
#pragma unroll
    for (int j = 0; j < 8; ++j) l[j] = g[j];
#endif
}

__device__ __forceinline__ void tile_copy_fence() {
#if ASYNC_LDS
    __builtin_amdgcn_s_wait_asynccnt(0);
#endif
    __syncthreads();
}

// -------------------------------------------------------------------------
// fp32 -> bf16 convert
// -------------------------------------------------------------------------
__global__ void convert_f32_bf16(const float* __restrict__ in,
                                 __bf16* __restrict__ out, int n) {
    int i = blockIdx.x * blockDim.x + threadIdx.x;
    if (i < n) out[i] = (__bf16)in[i];
}

// -------------------------------------------------------------------------
// Tiled WMMA GEMM: C[M,N] = alpha * A[M,K] @ (TRANSB ? B[N,K]^T : B[K,N])
// bf16 in, fp32 accumulate via v_wmma_f32_16x16x32_bf16 (wave32).
// Block tile 128x64, 256 threads = 8 waves, each wave 32x32 (4 WMMA tiles,
// 2 A frags x 2 B frags per k-step).
// skipUpper: skip blocks entirely above diagonal (S0 = Q K^T, causal).
// limitK:    k loop only up to row-block end (causal AV GEMM).
// OUTBF ? bf16 output (Cb) : fp32 output (Cf).
// -------------------------------------------------------------------------
template <int TRANSB, int OUTBF>
__global__ void gemm_wmma(const __bf16* __restrict__ A,
                          const __bf16* __restrict__ Bm,
                          float* __restrict__ Cf, __bf16* __restrict__ Cb,
                          int M, int N, int K, int lda, int ldb, int ldc,
                          long aStride, long bStride, long cStride,
                          float alpha, int skipUpper, int limitK) {
    __shared__ __bf16 As[128 * LDK];   // [row][k]
    __shared__ __bf16 Bs[64 * LDK];    // [col][k] (B staged transposed)

    int bz = blockIdx.z;
    A  += (long)bz * aStride;
    Bm += (long)bz * bStride;
    long coff = (long)bz * cStride;

    int r0 = blockIdx.y * 128;
    int c0 = blockIdx.x * 64;
    if (skipUpper && c0 > r0 + 127) return;
    int kEnd = limitK ? min(K, r0 + 128) : K;

    int tid  = threadIdx.x;
    int lane = tid & 31;
    int wid  = tid >> 5;
    int wr   = (wid >> 1) * 32;   // wave row base within block (0,32,64,96)
    int wc   = (wid & 1) * 32;    // wave col base within block (0,32)

    // cooperative-load thread assignments (all 16B-aligned transfers)
    int ar = tid >> 1;            // 0..127
    int ak = (tid & 1) * 16;      // 0,16
    int bn1 = tid >> 2;           // TRANSB=1: 0..63
    int bk1 = (tid & 3) * 8;      // 0,8,16,24
    int bk0 = tid >> 3;           // TRANSB=0: 0..31
    int bn0 = (tid & 7) * 8;      // 0..56

    v8f acc[4] = {};

    for (int k0 = 0; k0 < kEnd; k0 += 32) {
        // ---- stage A tile 128x32 (2x b128 per thread, contiguous)
        {
            const __bf16* src = A + (long)(r0 + ar) * lda + (k0 + ak);
            __bf16* dst = &As[ar * LDK + ak];
            g2l_b128(src, dst);
            g2l_b128(src + 8, dst + 8);
        }
        // ---- stage B tile into Bs[col][k]
        if (TRANSB) {
            const __bf16* src = Bm + (long)(c0 + bn1) * ldb + (k0 + bk1);
            g2l_b128(src, &Bs[bn1 * LDK + bk1]);
        } else {
            const __bf16* src = Bm + (long)(k0 + bk0) * ldb + (c0 + bn0);
#if !ASYNC_LDS
            __builtin_prefetch(src + 32 * ldb, 0, 0);
#endif
#pragma unroll
            for (int j = 0; j < 8; ++j) Bs[(bn0 + j) * LDK + bk0] = src[j];
        }
        tile_copy_fence();

        // ---- fragments in contiguous k order (k = 16h + s), same for A and B
        int rsel = lane & 15;
        int h    = lane >> 4;
        v16bf a0, a1;
        {
            // A 16x32 bf16: lanes 0-15 K 0..15, lanes 16-31 K 16..31
            const __bf16* p0 = &As[(wr + rsel) * LDK + h * 16];
            const __bf16* p1 = &As[(wr + 16 + rsel) * LDK + h * 16];
#pragma unroll
            for (int j = 0; j < 8; ++j) {
                a0[j] = p0[j]; a0[8 + j] = p0[8 + j];
                a1[j] = p1[j]; a1[8 + j] = p1[8 + j];
            }
        }
        v16bf b0, b1;
        {
            // B 32x16 bf16: lanes 0-15 K 0..15, lanes 16-31 K 16..31
            const __bf16* q0 = &Bs[(wc + rsel) * LDK + h * 16];
            const __bf16* q1 = &Bs[(wc + 16 + rsel) * LDK + h * 16];
#pragma unroll
            for (int j = 0; j < 16; ++j) { b0[j] = q0[j]; b1[j] = q1[j]; }
        }
        acc[0] = __builtin_amdgcn_wmma_f32_16x16x32_bf16(false, a0, false, b0,
                                                         (short)0, acc[0], false, false);
        acc[1] = __builtin_amdgcn_wmma_f32_16x16x32_bf16(false, a0, false, b1,
                                                         (short)0, acc[1], false, false);
        acc[2] = __builtin_amdgcn_wmma_f32_16x16x32_bf16(false, a1, false, b0,
                                                         (short)0, acc[2], false, false);
        acc[3] = __builtin_amdgcn_wmma_f32_16x16x32_bf16(false, a1, false, b1,
                                                         (short)0, acc[3], false, false);
        __syncthreads();
    }

    // ---- epilogue: C layout VGPR i -> (M = i + 8h, N = lane&15)
    int h  = lane >> 4;
    int cl = lane & 15;
#pragma unroll
    for (int t = 0; t < 4; ++t) {
        int tr = (t >> 1) * 16, tc = (t & 1) * 16;
#pragma unroll
        for (int i = 0; i < 8; ++i) {
            int row = r0 + wr + tr + i + 8 * h;
            int col = c0 + wc + tc + cl;
            float v = acc[t][i] * alpha;
            if (OUTBF) Cb[coff + (long)row * ldc + col] = (__bf16)v;
            else       Cf[coff + (long)row * ldc + col] = v;
        }
    }
}

// -------------------------------------------------------------------------
// Influence: positions = tanh(P + pos_b) + pbias[s]; w[tok,n] =
//   max(exp(-0.5*||pos-splat_n||^2/sc_n^2), 0.01).  One wave per token.
// -------------------------------------------------------------------------
__global__ void influence_kernel(const float* __restrict__ Pf,
                                 const float* __restrict__ pos_b,
                                 const float* __restrict__ pbias,
                                 const float* __restrict__ spos,
                                 const float* __restrict__ slog,
                                 float* __restrict__ W, int S, int nTok) {
    int wave = (blockIdx.x * blockDim.x + threadIdx.x) >> 5;
    int lane = threadIdx.x & 31;
    if (wave >= nTok) return;
    int s = wave % S;
    int e0 = lane, e1 = lane + 32;
    float p0 = tanhf(Pf[wave * 64 + e0] + pos_b[e0]) + pbias[s * 64 + e0];
    float p1 = tanhf(Pf[wave * 64 + e1] + pos_b[e1]) + pbias[s * 64 + e1];
#pragma unroll
    for (int n = 0; n < 16; ++n) {
        float d0 = p0 - spos[n * 64 + e0];
        float d1 = p1 - spos[n * 64 + e1];
        float d  = d0 * d0 + d1 * d1;
#pragma unroll
        for (int off = 16; off > 0; off >>= 1) d += __shfl_xor(d, off, 32);
        if (lane == 0) {
            float sc = __expf(slog[n]);
            sc = fminf(fmaxf(sc, 0.3f), 2.0f);
            float inf = fmaxf(__expf(-0.5f * d / (sc * sc)), 0.01f);
            W[wave * 16 + n] = inf;
        }
    }
}

// -------------------------------------------------------------------------
// Per-row online softmax stats (max m, sum Z) for all 16 splats at once.
// Logit_n(i,j) = w[i,n]*w[j,n]*S0[i,j], causal j<=i. One block per row.
// -------------------------------------------------------------------------
__global__ void row_stats_kernel(const float* __restrict__ S0,
                                 const float* __restrict__ W,
                                 float* __restrict__ Mst,
                                 float* __restrict__ Zst, int S) {
    int token = blockIdx.x;
    int bb = token / S, i = token % S;
    int tid = threadIdx.x;
    __shared__ float wsh[16];
    if (tid < 16) wsh[tid] = W[token * 16 + tid];
    __syncthreads();

    float m[16], z[16];
#pragma unroll
    for (int n = 0; n < 16; ++n) { m[n] = -1e30f; z[n] = 0.f; }

    const float* row = S0 + ((long)bb * S + i) * S;
    for (int j = tid; j <= i; j += 256) {
        float sv = row[j];
        const float* wj = &W[(bb * S + j) * 16];
#pragma unroll
        for (int n = 0; n < 16; ++n) {
            float t = wsh[n] * wj[n] * sv;
            if (t > m[n]) { z[n] = z[n] * __expf(m[n] - t) + 1.f; m[n] = t; }
            else          { z[n] += __expf(t - m[n]); }
        }
    }
    __shared__ float sm[256 * 16];
    __shared__ float sz[256 * 16];
#pragma unroll
    for (int n = 0; n < 16; ++n) { sm[tid * 16 + n] = m[n]; sz[tid * 16 + n] = z[n]; }
    __syncthreads();
    if (tid < 16) {
        int n = tid;
        float Mv = -1e30f, Zv = 0.f;
        for (int t = 0; t < 256; ++t) {
            float mm = sm[t * 16 + n], zz = sz[t * 16 + n];
            if (mm > Mv) { Zv = Zv * __expf(Mv - mm) + zz; Mv = mm; }
            else         { Zv += zz * __expf(mm - Mv); }
        }
        Mst[token * 16 + n] = Mv;
        Zst[token * 16 + n] = Zv;
    }
}

// -------------------------------------------------------------------------
// A_comb[i,j] = sum_n (g_n/16) * exp(w_i w_j s_ij - m_n)/Z_n  (j<=i, else 0)
// -> bf16. 16x64 blocks; skip blocks never read by the 128-row causal AV GEMM.
// -------------------------------------------------------------------------
__global__ void acomb_kernel(const float* __restrict__ S0,
                             const float* __restrict__ W,
                             const float* __restrict__ Mst,
                             const float* __restrict__ Zst,
                             const float* __restrict__ gates,
                             __bf16* __restrict__ Ac, int S) {
    int bb = blockIdx.z;
    int r0 = blockIdx.y * 16;
    int c0 = blockIdx.x * 64;
    if (c0 > (r0 | 127)) return;   // outside causal-read window of AV row block

    __shared__ float wi[16][16], mi[16][16], zinv[16][16], wj[64][16], gsh[16];
    int tid = threadIdx.x;
    {
        int r = tid >> 4, n = tid & 15;
        int ti = bb * S + r0 + r;
        wi[r][n]   = W[ti * 16 + n];
        mi[r][n]   = Mst[ti * 16 + n];
        zinv[r][n] = 1.f / Zst[ti * 16 + n];
    }
    for (int t = tid; t < 64 * 16; t += 256) {
        int r = t >> 4, n = t & 15;
        wj[r][n] = W[(bb * S + c0 + r) * 16 + n];
    }
    if (tid < 16) gsh[tid] = (1.f / (1.f + __expf(-gates[tid]))) * (1.f / 16.f);
    __syncthreads();

    int jc = tid & 63, rg = tid >> 6;
    for (int rr = rg; rr < 16; rr += 4) {
        int i = r0 + rr, j = c0 + jc;
        float val = 0.f;
        if (j <= i) {
            float sv = S0[((long)bb * S + i) * S + j];
#pragma unroll
            for (int n = 0; n < 16; ++n) {
                float t = wi[rr][n] * wj[jc][n] * sv;
                val += gsh[n] * __expf(t - mi[rr][n]) * zinv[rr][n];
            }
        }
        Ac[((long)bb * S + i) * S + j] = (__bf16)val;
    }
}

// -------------------------------------------------------------------------
// o = rw*x + (1-rw)*combined ; LayerNorm(D=1024, eps=1e-5). Block per token.
// -------------------------------------------------------------------------
__global__ void final_kernel(const float* __restrict__ X,
                             const float* __restrict__ comb,
                             const float* __restrict__ rw_p,
                             const float* __restrict__ lnw,
                             const float* __restrict__ lnb,
                             float* __restrict__ out, int D) {
    int token = blockIdx.x;
    int tid = threadIdx.x;
    float rw = 1.f / (1.f + __expf(-rw_p[0]));
    float o[4], sum = 0.f, sq = 0.f;
#pragma unroll
    for (int k = 0; k < 4; ++k) {
        int e = tid + k * 256;
        float v = rw * X[(long)token * D + e] + (1.f - rw) * comb[(long)token * D + e];
        o[k] = v; sum += v; sq += v * v;
    }
    __shared__ float s1[256], s2[256];
    s1[tid] = sum; s2[tid] = sq;
    __syncthreads();
    for (int st = 128; st > 0; st >>= 1) {
        if (tid < st) { s1[tid] += s1[tid + st]; s2[tid] += s2[tid + st]; }
        __syncthreads();
    }
    float mu  = s1[0] / (float)D;
    float var = s2[0] / (float)D - mu * mu;
    float inv = rsqrtf(var + 1e-5f);
#pragma unroll
    for (int k = 0; k < 4; ++k) {
        int e = tid + k * 256;
        out[(long)token * D + e] = (o[k] - mu) * inv * lnw[e] + lnb[e];
    }
}

// -------------------------------------------------------------------------
extern "C" void kernel_launch(void* const* d_in, const int* in_sizes, int n_in,
                              void* d_out, int out_size, void* d_ws, size_t ws_size,
                              hipStream_t stream) {
    (void)in_sizes; (void)n_in; (void)out_size; (void)ws_size;
    const int B = 2, S = 1024, D = 1024, E = 64;
    const int M = B * S;                 // 2048 tokens

    const float* X     = (const float*)d_in[0];
    const float* posW  = (const float*)d_in[1];
    const float* pos_b = (const float*)d_in[2];
    const float* pbias = (const float*)d_in[3];
    const float* spos  = (const float*)d_in[4];
    const float* slog  = (const float*)d_in[5];
    const float* qW    = (const float*)d_in[6];
    const float* kW    = (const float*)d_in[7];
    const float* vW    = (const float*)d_in[8];
    const float* gates = (const float*)d_in[9];
    const float* rw_p  = (const float*)d_in[10];
    const float* lnw   = (const float*)d_in[11];
    const float* lnb   = (const float*)d_in[12];
    float* out = (float*)d_out;

    // ---- workspace carve
    char* w = (char*)d_ws;
    auto carve = [&](size_t bytes) -> void* {
        void* p = (void*)w;
        w += (bytes + 255) & ~(size_t)255;
        return p;
    };
    __bf16* Xb    = (__bf16*)carve((size_t)M * D * 2);
    __bf16* qWb   = (__bf16*)carve((size_t)D * D * 2);
    __bf16* kWb   = (__bf16*)carve((size_t)D * D * 2);
    __bf16* vWb   = (__bf16*)carve((size_t)D * D * 2);
    __bf16* posWb = (__bf16*)carve((size_t)D * E * 2);
    __bf16* Qb    = (__bf16*)carve((size_t)M * D * 2);
    __bf16* Kb    = (__bf16*)carve((size_t)M * D * 2);
    __bf16* Vb    = (__bf16*)carve((size_t)M * D * 2);
    float*  Pf    = (float*)carve((size_t)M * E * 4);
    float*  Wf    = (float*)carve((size_t)M * 16 * 4);
    float*  S0    = (float*)carve((size_t)B * S * S * 4);
    float*  Mst   = (float*)carve((size_t)M * 16 * 4);
    float*  Zst   = (float*)carve((size_t)M * 16 * 4);
    __bf16* Ac    = (__bf16*)carve((size_t)B * S * S * 2);
    float*  Comb  = (float*)carve((size_t)M * D * 4);

    // ---- 1. converts to bf16
    {
        int n = M * D;
        convert_f32_bf16<<<(n + 255) / 256, 256, 0, stream>>>(X, Xb, n);
        n = D * D;
        convert_f32_bf16<<<(n + 255) / 256, 256, 0, stream>>>(qW, qWb, n);
        convert_f32_bf16<<<(n + 255) / 256, 256, 0, stream>>>(kW, kWb, n);
        convert_f32_bf16<<<(n + 255) / 256, 256, 0, stream>>>(vW, vWb, n);
        n = D * E;
        convert_f32_bf16<<<(n + 255) / 256, 256, 0, stream>>>(posW, posWb, n);
    }

    // ---- 2. Q/K/V GEMMs (shared weight, batch folded into M) -> bf16
    {
        dim3 grid(D / 64, M / 128, 1);
        gemm_wmma<0, 1><<<grid, 256, 0, stream>>>(Xb, qWb, nullptr, Qb,
            M, D, D, D, D, D, 0, 0, 0, 1.0f, 0, 0);
        gemm_wmma<0, 1><<<grid, 256, 0, stream>>>(Xb, kWb, nullptr, Kb,
            M, D, D, D, D, D, 0, 0, 0, 1.0f, 0, 0);
        gemm_wmma<0, 1><<<grid, 256, 0, stream>>>(Xb, vWb, nullptr, Vb,
            M, D, D, D, D, D, 0, 0, 0, 1.0f, 0, 0);
        // pos projection -> fp32
        dim3 gp(E / 64, M / 128, 1);
        gemm_wmma<0, 0><<<gp, 256, 0, stream>>>(Xb, posWb, Pf, nullptr,
            M, E, D, D, E, E, 0, 0, 0, 1.0f, 0, 0);
    }

    // ---- 3. influence weights
    influence_kernel<<<(M * 32 + 255) / 256, 256, 0, stream>>>(
        Pf, pos_b, pbias, spos, slog, Wf, S, M);

    // ---- 4. S0 = Q K^T / sqrt(D), causal block skip, fp32
    {
        dim3 grid(S / 64, S / 128, B);
        gemm_wmma<1, 0><<<grid, 256, 0, stream>>>(Qb, Kb, S0, nullptr,
            S, S, D, D, D, S,
            (long)S * D, (long)S * D, (long)S * S,
            1.0f / 32.0f /* 1/sqrt(1024) */, 1 /*skipUpper*/, 0);
    }

    // ---- 5. per-row softmax stats for all 16 splats
    row_stats_kernel<<<M, 256, 0, stream>>>(S0, Wf, Mst, Zst, S);

    // ---- 6. combined attention matrix (gated mean of 16 softmaxes) -> bf16
    {
        dim3 grid(S / 64, S / 16, B);
        acomb_kernel<<<grid, 256, 0, stream>>>(S0, Wf, Mst, Zst, gates, Ac, S);
    }

    // ---- 7. combined = A_comb @ V (causal K-limit), fp32
    {
        dim3 grid(D / 64, S / 128, B);
        gemm_wmma<0, 0><<<grid, 256, 0, stream>>>(Ac, Vb, Comb, nullptr,
            S, D, S, S, D, D,
            (long)S * S, (long)S * D, (long)S * D,
            1.0f, 0, 1 /*limitK*/);
    }

    // ---- 8. residual + LayerNorm
    final_kernel<<<M, 256, 0, stream>>>(X, Comb, rw_p, lnw, lnb, out, D);
}
